// ParameterizedKnowledgeStore_37864431681794
// MI455X (gfx1250) — matrix operand order repaired and checked by reading, not compiled, not run on hardware
//
#include <hip/hip_runtime.h>
#include <stdint.h>

#define K_SZ 32768
#define HID  1024
#define QD   512
#define NROWS 8192   // 4 * 2048

typedef __attribute__((ext_vector_type(16))) __bf16 v16bf;
typedef __attribute__((ext_vector_type(8)))  float  v8f;
typedef __attribute__((ext_vector_type(4)))  unsigned td_v4u;
typedef __attribute__((ext_vector_type(8)))  int      td_v8i;
typedef __attribute__((ext_vector_type(4)))  int      td_v4i;

#if __has_builtin(__builtin_amdgcn_tensor_load_to_lds) && __has_builtin(__builtin_amdgcn_s_wait_tensorcnt)
#define USE_TDM 1
#else
#define USE_TDM 0
#endif

union Frag { v16bf v; unsigned u[8]; };

__device__ __forceinline__ unsigned f2bf1(float f) {
  unsigned u = __builtin_bit_cast(unsigned, f);
  u += 0x7fffu + ((u >> 16) & 1u);           // round-to-nearest-even
  return u >> 16;
}
__device__ __forceinline__ unsigned packbf(float lo, float hi) {
  return f2bf1(lo) | (f2bf1(hi) << 16);
}

// ---------------- queries f32 -> bf16 ----------------
__global__ void cvt_q_kernel(const float* __restrict__ q,
                             unsigned short* __restrict__ qb) {
  int i = (blockIdx.x * 256 + threadIdx.x) * 2;
  float2 f = *(const float2*)(q + i);
  *(unsigned*)(qb + i) = packbf(f.x, f.y);
}

// ---------------- keys = KB @ Wk + bk  -> bf16 [K_SZ, QD] ----------------
__global__ void gemm_keys_kernel(const float* __restrict__ kb,
                                 const float* __restrict__ wk,
                                 const float* __restrict__ bias,
                                 unsigned short* __restrict__ keys) {
  const int lane = threadIdx.x & 31;
  const int wid  = threadIdx.x >> 5;
  const int tile = blockIdx.x * 8 + wid;
  const int NT = QD / 16;                 // 32 n-tiles
  const int mi = tile / NT, ni = tile % NT;
  const int m0 = mi * 16, n0 = ni * 16;
  const int ln = lane & 15, hb = lane >> 4;
  const int kbA = hb ? 16 : 0, kbB = hb ? 16 : 0;

  v8f acc = {};
  const float* arow = kb + (size_t)(m0 + ln) * HID;
  for (int k = 0; k < HID; k += 32) {
    Frag a, b;
#pragma unroll
    for (int v = 0; v < 8; ++v) {
      int k0 = k + kbA + 2 * v;
      float2 f = *(const float2*)(arow + k0);
      a.u[v] = packbf(f.x, f.y);
    }
#pragma unroll
    for (int v = 0; v < 8; ++v) {
      int k0 = k + kbB + 2 * v;
      float f0 = wk[(size_t)k0 * QD + (n0 + ln)];
      float f1 = wk[(size_t)(k0 + 1) * QD + (n0 + ln)];
      b.u[v] = packbf(f0, f1);
    }
    acc = __builtin_amdgcn_wmma_f32_16x16x32_bf16(false, a.v, false, b.v,
                                                  (short)0, acc, false, false);
  }
  float bv = bias[n0 + ln];
#pragma unroll
  for (int r = 0; r < 8; ++r) {
    int m = m0 + r + 8 * hb;
    keys[(size_t)m * QD + n0 + ln] = (unsigned short)f2bf1(acc[r] + bv);
  }
}

// ---------------- valuesT = (KB @ Wv + bv)^T -> bf16 [HID, K_SZ] ----------------
__global__ void gemm_values_kernel(const float* __restrict__ kb,
                                   const float* __restrict__ wv,
                                   const float* __restrict__ bias,
                                   unsigned short* __restrict__ valT) {
  const int lane = threadIdx.x & 31;
  const int wid  = threadIdx.x >> 5;
  const int tile = blockIdx.x * 8 + wid;
  const int NT = HID / 16;                // 64 n-tiles
  const int mi = tile / NT, ni = tile % NT;
  const int m0 = mi * 16, n0 = ni * 16;
  const int ln = lane & 15, hb = lane >> 4;
  const int kbA = hb ? 16 : 0, kbB = hb ? 16 : 0;

  v8f acc = {};
  const float* arow = kb + (size_t)(m0 + ln) * HID;
  for (int k = 0; k < HID; k += 32) {
    Frag a, b;
#pragma unroll
    for (int v = 0; v < 8; ++v) {
      int k0 = k + kbA + 2 * v;
      float2 f = *(const float2*)(arow + k0);
      a.u[v] = packbf(f.x, f.y);
    }
#pragma unroll
    for (int v = 0; v < 8; ++v) {
      int k0 = k + kbB + 2 * v;
      float f0 = wv[(size_t)k0 * HID + (n0 + ln)];
      float f1 = wv[(size_t)(k0 + 1) * HID + (n0 + ln)];
      b.u[v] = packbf(f0, f1);
    }
    acc = __builtin_amdgcn_wmma_f32_16x16x32_bf16(false, a.v, false, b.v,
                                                  (short)0, acc, false, false);
  }
  float bv = bias[n0 + ln];
#pragma unroll
  for (int r = 0; r < 8; ++r) {
    int m = m0 + r + 8 * hb;
    valT[(size_t)(n0 + ln) * K_SZ + m] = (unsigned short)f2bf1(acc[r] + bv);
  }
}

#if USE_TDM
// TDM: async DMA of one 64-row x 512-col bf16 key tile into LDS.
// D# per CDNA5 ISA: 2D tensor, data_size=2B, tile_dim0=512, tile_dim1=64,
// tensor_dim0_stride=512; pad 4 DWORDs every 256 DWORDs -> LDS row stride 520 u16.
__device__ __forceinline__ void tdm_load_keys(const unsigned short* gsrc, unsigned ldsAddr) {
  unsigned long long ga = (unsigned long long)(size_t)gsrc;
  td_v4u g0;
  g0[0] = 1u;                                              // count=1 (valid), user mode
  g0[1] = ldsAddr;                                         // lds_addr (bytes)
  g0[2] = (unsigned)(ga & 0xffffffffu);                    // global_addr[31:0]
  g0[3] = (unsigned)((ga >> 32) & 0x1ffffffu) | (2u << 30); // global_addr[56:32] | type=2
  td_v8i g1;
  g1[0] = (int)((1u << 16) | (1u << 20) | (7u << 22) | (3u << 25));
  //        data_size=2B | pad_enable | pad_interval=256DW | pad_amount=4DW
  g1[1] = (int)(512u << 16);        // tensor_dim0 = 512 (low16 in [63:48])
  g1[2] = (int)(64u << 16);         // tensor_dim1 = 64  (low16 in [95:80])
  g1[3] = (int)(512u << 16);        // tile_dim0 = 512   ([127:112])
  g1[4] = 64;                       // tile_dim1 = 64; tile_dim2 = 0
  g1[5] = 512;                      // tensor_dim0_stride = 512 (low32)
  g1[6] = 0;                        // stride hi / tensor_dim1_stride lo
  g1[7] = 0;
  td_v4i z4 = {0, 0, 0, 0};
  td_v8i z8;
  z8[0] = 0; z8[1] = 0; z8[2] = 0; z8[3] = 0;
  z8[4] = 0; z8[5] = 0; z8[6] = 0; z8[7] = 0;
  // 6-arg form (clang-23 / therock-10.0 headers): groups 0..3 + extra group + cpol
  __builtin_amdgcn_tensor_load_to_lds(g0, g1, z4, z4, z8, 0);
}
#endif

// ---------------- fused flash attention ----------------
// 256 blocks x 256 threads (8 waves). Block = 32 query rows, full H=1024.
__global__ void __launch_bounds__(256, 1)
flash_attn_kernel(const unsigned short* __restrict__ qb,    // [NROWS, QD]
                  const unsigned short* __restrict__ keys,  // [K_SZ, QD]
                  const unsigned short* __restrict__ valT,  // [HID, K_SZ]
                  float* __restrict__ out) {                // [NROWS, HID]
  const int lane = threadIdx.x & 31;
  const int wid  = threadIdx.x >> 5;
  const int ln = lane & 15, hb = lane >> 4;
  const int mi = wid >> 2;                 // 0..1 (16-row m-tile)
  const int ni = wid & 3;                  // 0..3 (16-col n-tile)
  const int rowBase = blockIdx.x * 32;
  const int kbA = hb ? 16 : 0, kbB = hb ? 16 : 0;
  const float L2E = 1.4426950408889634f;
  const int NTILE = K_SZ / 64;             // 512 knowledge tiles

  __shared__ unsigned short sQ[32][QD + 8];
  __shared__ unsigned short sK[2][64][QD + 8];   // double-buffered key tiles
  __shared__ unsigned short sP[32][64];
  __shared__ float sRunMax[32], sRunSum[32], sScale[32], sNewMax[32];
  __shared__ float sTileMax[2][4][16];

  { // stage Q block (reused 512x)
    const unsigned* src = (const unsigned*)(qb + (size_t)rowBase * QD);
    for (int i = threadIdx.x; i < 32 * (QD / 2); i += 256) {
      int r = i / (QD / 2), c = i % (QD / 2);
      *(unsigned*)&sQ[r][c * 2] = src[(size_t)r * (QD / 2) + c];
    }
    if (threadIdx.x < 32) { sRunMax[threadIdx.x] = -3.0e38f; sRunSum[threadIdx.x] = 0.f; }
  }

#if USE_TDM
  const unsigned sKbase = (unsigned)(size_t)&sK[0][0][0];
  const unsigned sKstride = (unsigned)(64 * (QD + 8) * 2);   // bytes per buffer
  if (wid == 0) tdm_load_keys(keys, sKbase);                 // prefetch tile 0
#endif
  __syncthreads();

  v8f acc[16];
#pragma unroll
  for (int t = 0; t < 16; ++t) acc[t] = (v8f){};

  for (int it = 0; it < NTILE; ++it) {
    const int buf = it & 1;
#if USE_TDM
    if (wid == 0) {
      if (it + 1 < NTILE) {   // issue next tile, then wait until current landed
        tdm_load_keys(keys + (size_t)(it + 1) * 64 * QD, sKbase + (buf ^ 1) * sKstride);
        __builtin_amdgcn_s_wait_tensorcnt(1);
      } else {
        __builtin_amdgcn_s_wait_tensorcnt(0);
      }
    }
    __syncthreads();
#else
    { // cooperative synchronous staging fallback
      const uint4* src = (const uint4*)(keys + (size_t)it * 64 * QD);
      for (int i = threadIdx.x; i < 64 * 64; i += 256) {
        int r = i >> 6, c = i & 63;
        *(uint4*)&sK[buf][r][c * 8] = src[(size_t)r * 64 + c];
      }
    }
    __syncthreads();
#endif

    // ---- score tile: 16x16 per wave, K=512, A from sQ, B from sK (ds_load)
    v8f s = {};
    {
      const unsigned short* kr = &sK[buf][ni * 16 + ln][0];
      const unsigned short* qr = &sQ[mi * 16 + ln][0];
      for (int k = 0; k < QD; k += 32) {
        Frag a, b;
#pragma unroll
        for (int v = 0; v < 8; ++v) {
          int k0 = k + kbA + 2 * v;
          a.u[v] = *(const unsigned*)(qr + k0);
        }
#pragma unroll
        for (int v = 0; v < 8; ++v) {
          int k0 = k + kbB + 2 * v;
          b.u[v] = *(const unsigned*)(kr + k0);
        }
        s = __builtin_amdgcn_wmma_f32_16x16x32_bf16(false, a.v, false, b.v,
                                                    (short)0, s, false, false);
      }
    }
    // ---- per-row max inside tile (rows span 16 lanes of each half)
    float tmax[8];
#pragma unroll
    for (int r = 0; r < 8; ++r) {
      float v = s[r];
      v = fmaxf(v, __shfl_xor(v, 1));
      v = fmaxf(v, __shfl_xor(v, 2));
      v = fmaxf(v, __shfl_xor(v, 4));
      v = fmaxf(v, __shfl_xor(v, 8));
      tmax[r] = v;
    }
    if (ln == 0) {
#pragma unroll
      for (int r = 0; r < 8; ++r) sTileMax[mi][ni][r + 8 * hb] = tmax[r];
    }
    __syncthreads();
    // ---- online softmax stats (one wave, 32 rows)
    if (wid == 0) {
      int row = lane;
      float tm = fmaxf(fmaxf(sTileMax[row >> 4][0][row & 15], sTileMax[row >> 4][1][row & 15]),
                       fmaxf(sTileMax[row >> 4][2][row & 15], sTileMax[row >> 4][3][row & 15]));
      float om = sRunMax[row];
      float nm = fmaxf(om, tm);
      float sc = __builtin_exp2f((om - nm) * L2E);
      sNewMax[row] = nm; sScale[row] = sc; sRunMax[row] = nm;
      sRunSum[row] *= sc;
    }
    __syncthreads();
    // ---- P = exp(S - newmax), write bf16 tile + accumulate row sums
#pragma unroll
    for (int r = 0; r < 8; ++r) {
      int mloc = mi * 16 + r + 8 * hb;
      float p = __builtin_exp2f((s[r] - sNewMax[mloc]) * L2E);
      sP[mloc][ni * 16 + ln] = (unsigned short)f2bf1(p);
      float ps = p;
      ps += __shfl_xor(ps, 1);
      ps += __shfl_xor(ps, 2);
      ps += __shfl_xor(ps, 4);
      ps += __shfl_xor(ps, 8);
      if (ln == 0) atomicAdd(&sRunSum[mloc], ps);   // ds_add_f32
    }
    __syncthreads();
    // ---- rescale accumulators by per-row scale
    float sc[8];
#pragma unroll
    for (int r = 0; r < 8; ++r) sc[r] = sScale[mi * 16 + r + 8 * hb];
#pragma unroll
    for (int t = 0; t < 16; ++t)
#pragma unroll
      for (int r = 0; r < 8; ++r) acc[t][r] *= sc[r];
    // ---- PV: acc += P(16x64) x V(64 x 256-col slice); valuesT L2-resident
#pragma unroll
    for (int kk = 0; kk < 2; ++kk) {
      Frag a;
      const unsigned short* pr = &sP[mi * 16 + ln][kk * 32];
#pragma unroll
      for (int v = 0; v < 8; ++v) {
        int k0 = kbA + 2 * v;
        a.u[v] = *(const unsigned*)(pr + k0);
      }
#pragma unroll
      for (int t = 0; t < 16; ++t) {
        int ht = (wid & 3) + 4 * t;       // h-tile 0..63
        const unsigned short* vr = valT + (size_t)(ht * 16 + ln) * K_SZ + (size_t)it * 64 + kk * 32;
        Frag b;
#pragma unroll
        for (int v = 0; v < 8; ++v) {
          int k0 = kbB + 2 * v;
          b.u[v] = *(const unsigned*)(vr + k0);
        }
        acc[t] = __builtin_amdgcn_wmma_f32_16x16x32_bf16(false, a.v, false, b.v,
                                                         (short)0, acc[t], false, false);
      }
    }
    __syncthreads();
  }

  // ---- normalize by row sum and store f32
  float inv[8];
#pragma unroll
  for (int r = 0; r < 8; ++r) inv[r] = 1.0f / sRunSum[mi * 16 + r + 8 * hb];
#pragma unroll
  for (int t = 0; t < 16; ++t) {
    int ht = (wid & 3) + 4 * t;
    int h = ht * 16 + ln;
#pragma unroll
    for (int r = 0; r < 8; ++r) {
      int row = rowBase + mi * 16 + r + 8 * hb;
      out[(size_t)row * HID + h] = acc[t][r] * inv[r];
    }
  }
}

extern "C" void kernel_launch(void* const* d_in, const int* in_sizes, int n_in,
                              void* d_out, int out_size, void* d_ws, size_t ws_size,
                              hipStream_t stream) {
  (void)in_sizes; (void)n_in; (void)out_size; (void)ws_size;
  const float* q  = (const float*)d_in[0];   // [4,2048,512]
  const float* kb = (const float*)d_in[1];   // [32768,1024]
  const float* Wk = (const float*)d_in[2];   // [1024,512]
  const float* bk = (const float*)d_in[3];   // [512]
  const float* Wv = (const float*)d_in[4];   // [1024,1024]
  const float* bv = (const float*)d_in[5];   // [1024]
  float* out = (float*)d_out;

  char* ws = (char*)d_ws;
  unsigned short* qb   = (unsigned short*)ws;                         //  8 MiB
  unsigned short* keys = (unsigned short*)(ws + (size_t)(8u << 20));  // 32 MiB
  unsigned short* valT = (unsigned short*)(ws + (size_t)(40u << 20)); // 64 MiB

  cvt_q_kernel<<<(NROWS * QD / 2) / 256, 256, 0, stream>>>(q, qb);
  gemm_keys_kernel<<<(K_SZ / 16) * (QD / 16) / 8, 256, 0, stream>>>(kb, Wk, bk, keys);
  gemm_values_kernel<<<(K_SZ / 16) * (HID / 16) / 8, 256, 0, stream>>>(kb, Wv, bv, valT);
  flash_attn_kernel<<<NROWS / 32, 256, 0, stream>>>(qb, keys, valT, out);
}
